// Attention_12695923327433
// MI455X (gfx1250) — compile-verified, matrix operand order checked by reading, not run on hardware
//
#include <hip/hip_runtime.h>
#include <hip/hip_bf16.h>
#include <math.h>

// ---------------------------------------------------------------------------
// CDNA5 (gfx1250) WMMA types / helpers
// ---------------------------------------------------------------------------
typedef __attribute__((ext_vector_type(16))) __bf16 v16bf;
typedef __attribute__((ext_vector_type(8)))  float  v8f;

// Native conversion: lets the backend use hardware f32->bf16 cvt (RNE) instead
// of integer-emulated rounding.
__device__ __forceinline__ __bf16 f2bf(float f) { return (__bf16)f; }

// A/B 16-bit fragment K-pattern for v_wmma_*_16x16x32 (wave32):
//   lane = l0 + 16*hi (l0 = row of A / col of B), VGPR v holds packed K pair:
//   k = (v>>2)*16 + hi*8 + (v&3)*2  and k+1.
__device__ __forceinline__ int frag_kbase(int v, int hi) {
    return ((v >> 2) << 4) + (hi << 3) + ((v & 3) << 1);
}

// ---------------------------------------------------------------------------
// gfx1250 async global->LDS copies (ASYNCcnt pipe), with sync fallback.
// Builtin signature (from compiler diagnostic): param0 = vector-of-int ptr in
// addrspace(1), param1 = LDS-side ptr, then imm offset + cpol.
// ---------------------------------------------------------------------------
#define AS1 __attribute__((address_space(1)))
#define AS3 __attribute__((address_space(3)))
typedef int v4i __attribute__((vector_size(16)));
typedef int v2i __attribute__((vector_size(8)));

#if __has_builtin(__builtin_amdgcn_global_load_async_to_lds_b128) && \
    __has_builtin(__builtin_amdgcn_global_load_async_to_lds_b64)  && \
    __has_builtin(__builtin_amdgcn_s_wait_asynccnt)
#define USE_ASYNC_LDS 1
#else
#define USE_ASYNC_LDS 0
#endif

__device__ __forceinline__ void copy16_g2l(const float* g, float* l) {
#if USE_ASYNC_LDS
    __builtin_amdgcn_global_load_async_to_lds_b128((AS1 v4i*)g, (AS3 v4i*)l, 0, 0);
#else
    *(float4*)l = *(const float4*)g;
#endif
}
__device__ __forceinline__ void copy8_g2l(const float* g, float* l) {
#if USE_ASYNC_LDS
    __builtin_amdgcn_global_load_async_to_lds_b64((AS1 v2i*)g, (AS3 v2i*)l, 0, 0);
#else
    *(float2*)l = *(const float2*)g;
#endif
}
__device__ __forceinline__ void async_wait0() {
#if USE_ASYNC_LDS
    __builtin_amdgcn_s_wait_asynccnt(0);
#endif
}

#define DIMC   384
#define HEADS  8
#define CPH    48       // channels per head
#define BATCH  4
#define HW     16384    // 128*128
#define IMH    128
#define IMW    128

// ---------------------------------------------------------------------------
// Dense GEMM  Y[b,M,N] (+)= W[M,K] * X[b,K,N]   via bf16 WMMA.
//   block = 256 threads (8 waves); block tile = 64(M) x 128(N); K step 32.
//   Each wave owns one 16-wide N slice and 4 stacked M accumulators:
//   B-fragment built once per K-step, reused by 4 WMMAs. Tiles staged via
//   async-to-LDS (ASYNCcnt) with per-wave wait before the workgroup barrier.
//   Used for qkv 1x1 conv (M=1152,K=384) and proj 1x1 conv (M=384,K=384,acc).
// ---------------------------------------------------------------------------
__global__ __launch_bounds__(256)
void gemm_wmma_bf16(const float* __restrict__ Wm, const float* __restrict__ X,
                    float* __restrict__ Y, int M, int K, int N, int accumulate)
{
    __shared__ float lds_a[64 * 32];   // 8 KB
    __shared__ float lds_b[32 * 128];  // 16 KB

    const int tid  = threadIdx.x;
    const int wave = tid >> 5;
    const int lane = tid & 31;
    const int l0   = lane & 15;
    const int hi   = lane >> 4;
    const int m0   = blockIdx.y * 64;
    const int n0   = blockIdx.x * 128;
    const size_t xb = (size_t)blockIdx.z * K * N;
    const size_t yb = (size_t)blockIdx.z * M * N;

    int kb[8];
#pragma unroll
    for (int v = 0; v < 8; ++v) kb[v] = frag_kbase(v, hi);

    v8f acc[4] = {};

    for (int k0 = 0; k0 < K; k0 += 32) {
        // stage A tile (64x32 f32, 2048 elem): thread -> 8 consecutive floats
        {
            int r = tid >> 2, c = (tid & 3) << 3;
            const float* g = &Wm[(size_t)(m0 + r) * K + k0 + c];
            float* l = &lds_a[r * 32 + c];
            copy16_g2l(g, l);
            copy16_g2l(g + 4, l + 4);
        }
        // stage B tile (32x128 f32, 4096 elem): thread -> 4 float4
#pragma unroll
        for (int j = 0; j < 4; ++j) {
            int f  = tid + j * 256;
            int r  = f >> 5;
            int c4 = (f & 31) << 2;
            copy16_g2l(&X[xb + (size_t)(k0 + r) * N + n0 + c4],
                       &lds_b[r * 128 + c4]);
        }
        if (k0 + 32 < K)  // speculative prefetch of next B tile -> global_prefetch_b8
            __builtin_prefetch(&X[xb + (size_t)(k0 + 32) * N + n0 + (tid >> 1)], 0, 0);
        async_wait0();       // this wave's async copies landed in LDS
        __syncthreads();     // all waves' copies visible

        v16bf b;
#pragma unroll
        for (int v = 0; v < 8; ++v) {
            int kk = kb[v];
            b[2 * v]     = f2bf(lds_b[kk * 128 + wave * 16 + l0]);
            b[2 * v + 1] = f2bf(lds_b[(kk + 1) * 128 + wave * 16 + l0]);
        }
#pragma unroll
        for (int mt = 0; mt < 4; ++mt) {
            v16bf a;
#pragma unroll
            for (int v = 0; v < 8; ++v) {
                int kk = kb[v];
                a[2 * v]     = f2bf(lds_a[(mt * 16 + l0) * 32 + kk]);
                a[2 * v + 1] = f2bf(lds_a[(mt * 16 + l0) * 32 + kk + 1]);
            }
            acc[mt] = __builtin_amdgcn_wmma_f32_16x16x32_bf16(false, a, false, b,
                                                              (short)0, acc[mt],
                                                              false, false);
        }
        __syncthreads();
    }

    // C/D layout: VGPR r -> row (+ r + 8*hi), col n0 + wave*16 + l0
#pragma unroll
    for (int mt = 0; mt < 4; ++mt) {
#pragma unroll
        for (int r = 0; r < 8; ++r) {
            size_t idx = yb + (size_t)(m0 + mt * 16 + r + hi * 8) * N
                            + n0 + wave * 16 + l0;
            float o = acc[mt][r];
            if (accumulate) o += Y[idx];
            Y[idx] = o;
        }
    }
}

// ---------------------------------------------------------------------------
// Generic depthwise 3x3 SAME conv, optional exact GELU on output.
//   in element:  in[((b*inCtot + inCoff + c) * H + y) * W + x]
//   out element: out[((b*C + c) * H + y) * W + x]
// ---------------------------------------------------------------------------
__global__ void dwconv3x3(const float* __restrict__ in, const float* __restrict__ wgt,
                          float* __restrict__ out, int C, int H, int W, int B,
                          int inCtot, int inCoff, int gelu)
{
    size_t idx = (size_t)blockIdx.x * blockDim.x + threadIdx.x;
    size_t total = (size_t)B * C * H * W;
    if (idx >= total) return;
    int x = (int)(idx % W);
    int y = (int)((idx / W) % H);
    int c = (int)((idx / ((size_t)W * H)) % C);
    int b = (int)(idx / ((size_t)W * H * C));

    const float* ip = in + ((size_t)b * inCtot + inCoff + c) * H * W;
    const float* wp = wgt + (size_t)c * 9;
    float s = 0.f;
#pragma unroll
    for (int dy = -1; dy <= 1; ++dy) {
        int yy = y + dy;
        if (yy < 0 || yy >= H) continue;
#pragma unroll
        for (int dx = -1; dx <= 1; ++dx) {
            int xx = x + dx;
            if (xx < 0 || xx >= W) continue;
            s += ip[(size_t)yy * W + xx] * wp[(dy + 1) * 3 + (dx + 1)];
        }
    }
    if (gelu) s = 0.5f * s * (1.0f + erff(s * 0.70710678118654752f));
    out[idx] = s;
}

// ---------------------------------------------------------------------------
// Inverse L2 norms of q and k rows: inv[b][a][c], a in {0:q, 1:k}
// ---------------------------------------------------------------------------
__global__ __launch_bounds__(256)
void l2_invnorm(const float* __restrict__ qkv, float* __restrict__ inv)
{
    __shared__ float red[256];
    int b = blockIdx.y;
    int a = blockIdx.x / DIMC;       // 0=q, 1=k
    int c = blockIdx.x % DIMC;
    const float* row = qkv + ((size_t)b * 3 * DIMC + a * DIMC + c) * HW;
    float s = 0.f;
    for (int i = threadIdx.x; i < HW; i += 256) { float v = row[i]; s += v * v; }
    red[threadIdx.x] = s;
    __syncthreads();
    for (int w = 128; w > 0; w >>= 1) {
        if (threadIdx.x < w) red[threadIdx.x] += red[threadIdx.x + w];
        __syncthreads();
    }
    if (threadIdx.x == 0)
        inv[(size_t)b * 2 * DIMC + a * DIMC + c] = 1.0f / fmaxf(sqrtf(red[0]), 1e-12f);
}

// ---------------------------------------------------------------------------
// attn0[bh][48][48] = temperature[h] * (q_norm . k_norm^T)
//   one wave per 16x16 tile; K = 16384 reduced in 32-steps of bf16 WMMA,
//   l2-norm scale fused into the f32->bf16 operand conversion.
// ---------------------------------------------------------------------------
__global__ __launch_bounds__(32)
void attn_scores_wmma(const float* __restrict__ qkv, const float* __restrict__ inv,
                      const float* __restrict__ temperature, float* __restrict__ scores)
{
    const int lane = threadIdx.x;
    const int l0 = lane & 15, hi = lane >> 4;
    const int m0 = blockIdx.x * 16;
    const int n0 = blockIdx.y * 16;
    const int bh = blockIdx.z;
    const int b = bh >> 3, h = bh & 7;

    const float* qrow = qkv + ((size_t)b * 3 * DIMC + h * CPH + m0 + l0) * HW;
    const float* krow = qkv + ((size_t)b * 3 * DIMC + DIMC + h * CPH + n0 + l0) * HW;
    const float iq = inv[(size_t)b * 2 * DIMC + h * CPH + m0 + l0];
    const float ik = inv[(size_t)b * 2 * DIMC + DIMC + h * CPH + n0 + l0];

    int kb[8];
#pragma unroll
    for (int v = 0; v < 8; ++v) kb[v] = frag_kbase(v, hi);

    v8f acc = {};
    for (int k0 = 0; k0 < HW; k0 += 32) {
        v16bf a, bfr;
#pragma unroll
        for (int v = 0; v < 8; ++v) {
            float2 qa = *(const float2*)&qrow[k0 + kb[v]];
            float2 ka = *(const float2*)&krow[k0 + kb[v]];
            a[2 * v]       = f2bf(qa.x * iq);
            a[2 * v + 1]   = f2bf(qa.y * iq);
            bfr[2 * v]     = f2bf(ka.x * ik);
            bfr[2 * v + 1] = f2bf(ka.y * ik);
        }
        acc = __builtin_amdgcn_wmma_f32_16x16x32_bf16(false, a, false, bfr,
                                                      (short)0, acc, false, false);
    }
    const float t = temperature[h];
#pragma unroll
    for (int r = 0; r < 8; ++r)
        scores[(size_t)bh * CPH * CPH + (size_t)(m0 + r + hi * 8) * CPH + n0 + l0] = acc[r] * t;
}

// ---------------------------------------------------------------------------
// Blended (plain + masked) softmax over each 48-wide row, in place.
// ---------------------------------------------------------------------------
__global__ void softmax_blend(float* __restrict__ scores, const int* __restrict__ mask,
                              const float* __restrict__ w_blend)
{
    int gid = blockIdx.x * blockDim.x + threadIdx.x;
    if (gid >= BATCH * HEADS * CPH) return;
    int row = gid % CPH;
    int bh  = gid / CPH;
    int h   = bh & 7;
    float* s = scores + (size_t)bh * CPH * CPH + (size_t)row * CPH;
    const int* m = mask + ((size_t)h * CPH + row) * CPH;

    float wa = w_blend[0], wb = w_blend[1];
    float wm = fmaxf(wa, wb);
    float ea = expf(wa - wm), eb = expf(wb - wm);
    float w0 = ea / (ea + eb), w1 = eb / (ea + eb);

    float buf0[CPH], buf1[CPH];
    float mx0 = -INFINITY, mx1 = -INFINITY;
    for (int i = 0; i < CPH; ++i) {
        float v = s[i];
        buf0[i] = v;
        buf1[i] = (m[i] == 0) ? -1e9f : v;
        mx0 = fmaxf(mx0, buf0[i]);
        mx1 = fmaxf(mx1, buf1[i]);
    }
    float s0 = 0.f, s1 = 0.f;
    for (int i = 0; i < CPH; ++i) {
        buf0[i] = expf(buf0[i] - mx0); s0 += buf0[i];
        buf1[i] = expf(buf1[i] - mx1); s1 += buf1[i];
    }
    float r0 = w0 / s0, r1 = w1 / s1;
    for (int i = 0; i < CPH; ++i) s[i] = buf0[i] * r0 + buf1[i] * r1;
}

// ---------------------------------------------------------------------------
// out[b, h*48+m, n] = sum_d attn[bh][m][d] * v[bh][d][n]
//   M=48, K=48 (zero-padded to 64), N=16384. One wave per 16x16 tile.
// ---------------------------------------------------------------------------
__global__ __launch_bounds__(32)
void attn_apply_wmma(const float* __restrict__ attn, const float* __restrict__ qkv,
                     float* __restrict__ out)
{
    const int lane = threadIdx.x;
    const int l0 = lane & 15, hi = lane >> 4;
    const int n0 = blockIdx.x * 16;
    const int m0 = blockIdx.y * 16;
    const int bh = blockIdx.z;
    const int b = bh >> 3, h = bh & 7;

    const float* A = attn + (size_t)bh * CPH * CPH;
    const float* V = qkv + ((size_t)b * 3 * DIMC + 2 * DIMC + h * CPH) * HW;

    int kb[8];
#pragma unroll
    for (int v = 0; v < 8; ++v) kb[v] = frag_kbase(v, hi);

    v8f acc = {};
#pragma unroll
    for (int s = 0; s < 2; ++s) {
        v16bf a, bb;
#pragma unroll
        for (int v = 0; v < 8; ++v) {
            int k = s * 32 + kb[v];
            float a0 = (k < CPH)     ? A[(size_t)(m0 + l0) * CPH + k]     : 0.f;
            float a1 = (k + 1 < CPH) ? A[(size_t)(m0 + l0) * CPH + k + 1] : 0.f;
            float b0 = (k < CPH)     ? V[(size_t)k * HW + n0 + l0]        : 0.f;
            float b1 = (k + 1 < CPH) ? V[(size_t)(k + 1) * HW + n0 + l0]  : 0.f;
            a[2 * v] = f2bf(a0);  a[2 * v + 1] = f2bf(a1);
            bb[2 * v] = f2bf(b0); bb[2 * v + 1] = f2bf(b1);
        }
        acc = __builtin_amdgcn_wmma_f32_16x16x32_bf16(false, a, false, bb,
                                                      (short)0, acc, false, false);
    }
#pragma unroll
    for (int r = 0; r < 8; ++r)
        out[((size_t)b * DIMC + h * CPH + m0 + r + hi * 8) * HW + n0 + l0] = acc[r];
}

// ---------------------------------------------------------------------------
// Host-side orchestration
// ---------------------------------------------------------------------------
extern "C" void kernel_launch(void* const* d_in, const int* in_sizes, int n_in,
                              void* d_out, int out_size, void* d_ws, size_t ws_size,
                              hipStream_t stream)
{
    (void)in_sizes; (void)n_in; (void)out_size; (void)ws_size;
    const float* x      = (const float*)d_in[0];  // (4,384,128,128)
    const int*   mask   = (const int*)  d_in[1];  // (1,8,48,48)
    const float* qkv_w  = (const float*)d_in[2];  // (1152,384,1,1)
    const float* dw_w   = (const float*)d_in[3];  // (1152,1,3,3)
    const float* proj_w = (const float*)d_in[4];  // (384,384,1,1)
    const float* temp   = (const float*)d_in[5];  // (8,1,1)
    const float* wblend = (const float*)d_in[6];  // (2,)
    const float* pos_w1 = (const float*)d_in[7];  // (384,1,3,3)
    const float* pos_w2 = (const float*)d_in[8];  // (384,1,3,3)
    float*       out    = (float*)d_out;          // (4,384,128,128)

    char* ws = (char*)d_ws;
    const size_t SZ_QKV  = (size_t)BATCH * 3 * DIMC * HW * sizeof(float); // ~302 MB
    const size_t SZ_FMAP = (size_t)BATCH * DIMC * HW * sizeof(float);     // ~100 MB
    float* qkv_raw  = (float*)ws;                       // [0, 302MB) -- dead after dwconv
    float* qkv_dw   = (float*)(ws + SZ_QKV);            // [302MB, 604MB)
    float* attn_out = qkv_raw;                          // reuse [0, 100MB)
    float* pos_tmp  = (float*)(ws + SZ_FMAP);           // reuse [100MB, 200MB)
    float* scores   = (float*)(ws + 2 * SZ_FMAP);       // reuse, ~295 KB
    float* invn     = (float*)(ws + 2 * SZ_FMAP + (1u << 21)); // 2 MB past scores

    // 1) qkv = conv1x1(x) as bf16-WMMA GEMM: Y[b,1152,16384] = W[1152,384]·X[b,384,16384]
    gemm_wmma_bf16<<<dim3(HW / 128, 3 * DIMC / 64, BATCH), 256, 0, stream>>>(
        qkv_w, x, qkv_raw, 3 * DIMC, DIMC, HW, 0);

    // 2) depthwise 3x3 over all 1152 channels
    {
        size_t total = (size_t)BATCH * 3 * DIMC * HW;
        dwconv3x3<<<dim3((unsigned)((total + 255) / 256)), 256, 0, stream>>>(
            qkv_raw, dw_w, qkv_dw, 3 * DIMC, IMH, IMW, BATCH, 3 * DIMC, 0, 0);
    }

    // 3) inverse L2 norms of q and k rows
    l2_invnorm<<<dim3(2 * DIMC, BATCH), 256, 0, stream>>>(qkv_dw, invn);

    // 4) attn0 = temperature * (q_norm · k_norm^T) : WMMA over K=16384
    attn_scores_wmma<<<dim3(3, 3, BATCH * HEADS), 32, 0, stream>>>(
        qkv_dw, invn, temp, scores);

    // 5) blended softmax (plain + masked), in place on scores
    softmax_blend<<<dim3((BATCH * HEADS * CPH + 255) / 256), 256, 0, stream>>>(
        scores, mask, wblend);

    // 6) out_heads = attn · v : WMMA, K=48 padded to 64
    attn_apply_wmma<<<dim3(HW / 16, 3, BATCH * HEADS), 32, 0, stream>>>(
        scores, qkv_dw, attn_out);

    // 7) positional path: dw3x3 + exact GELU, then dw3x3 -> writes pos into d_out
    {
        size_t total = (size_t)BATCH * DIMC * HW;
        dwconv3x3<<<dim3((unsigned)((total + 255) / 256)), 256, 0, stream>>>(
            qkv_dw, pos_w1, pos_tmp, DIMC, IMH, IMW, BATCH, 3 * DIMC, 2 * DIMC, 1);
        dwconv3x3<<<dim3((unsigned)((total + 255) / 256)), 256, 0, stream>>>(
            pos_tmp, pos_w2, out, DIMC, IMH, IMW, BATCH, DIMC, 0, 0);
    }

    // 8) out += conv1x1(attn_out, proj_w) as WMMA GEMM with accumulate
    gemm_wmma_bf16<<<dim3(HW / 128, DIMC / 64, BATCH), 256, 0, stream>>>(
        proj_w, attn_out, out, DIMC, DIMC, HW, 1);
}
